// SelfAggregation_55688545960041
// MI455X (gfx1250) — compile-verified, matrix operand order checked by reading, not run on hardware
//
#include <hip/hip_runtime.h>
#include <hip/hip_bf16.h>

// PointNet++ segmentation forward on MI455X (gfx1250).
// All MLP layers run as bf16 WMMA GEMMs (v_wmma_f32_16x16x32_bf16, wave32,
// 16x64 tile per wave). Irregular stages (FPS / ball-query / kNN / pooling /
// concat+pack) are scalar kernels that also handle f32<->bf16 conversion and
// K-padding to multiples of 32.

typedef unsigned short bf16_t;
typedef __attribute__((ext_vector_type(16))) __bf16 v16bf;
typedef __attribute__((ext_vector_type(8)))  float  v8f;

union FragBF { unsigned int u[8]; v16bf v; };

__device__ __forceinline__ bf16_t f2bf(float f) {
  unsigned int u = __float_as_uint(f);
  u += 0x7fffu + ((u >> 16) & 1u);          // round-to-nearest-even
  return (bf16_t)(u >> 16);
}
__device__ __forceinline__ float bf2f(bf16_t h) {
  return __uint_as_float(((unsigned int)h) << 16);
}

// K index base for VGPR r, half h in the 16-bit 16x32 A-matrix layout
// (ISA 7.12.2): lanes 0-15 V0..V3 = K{0,1..6,7}, V4..V7 = K{16,17..22,23};
// lanes 16-31 shift K by +8.
__device__ __forceinline__ int kmap(int r, int h) {
  return (r < 4 ? 2 * r : 16 + 2 * (r - 4)) + 8 * h;
}

// ---------------------------------------------------------------------------
// GEMM: out = relu(A[Mp x Kp] * W + bias), W given pre-transposed bf16 [N][Kp].
// One wave -> 16(M) x 64(N) tile, 4 f32 accumulators, K-loop step 32.
// ---------------------------------------------------------------------------
__global__ void gemm_bf16_kernel(const bf16_t* __restrict__ A,
                                 const bf16_t* __restrict__ Wt,
                                 const float*  __restrict__ bias,
                                 bf16_t* __restrict__ outBf,
                                 float*  __restrict__ outF,
                                 int Mp, int Kp, int N) {
  int gtid = blockIdx.x * blockDim.x + threadIdx.x;
  int wave = gtid >> 5;
  int lane = threadIdx.x & 31;
  int tilesN = N >> 6;
  int tilesM = Mp >> 4;
  if (wave >= tilesM * tilesN) return;          // whole-wave exit: EXEC stays all-ones
  int tm = wave / tilesN, tn = wave % tilesN;
  int m0 = tm << 4, n0 = tn << 6;
  int ln16 = lane & 15;
  int h    = lane >> 4;

  v8f acc[4];
#pragma unroll
  for (int t = 0; t < 4; ++t)
#pragma unroll
    for (int r = 0; r < 8; ++r) acc[t][r] = 0.0f;

  const bf16_t* Arow = A + (size_t)(m0 + ln16) * Kp;
  for (int k0 = 0; k0 < Kp; k0 += 32) {
    FragBF a;
#pragma unroll
    for (int r = 0; r < 8; ++r)
      a.u[r] = *(const unsigned int*)(Arow + k0 + kmap(r, h));
#pragma unroll
    for (int t = 0; t < 4; ++t) {
      const bf16_t* Wrow = Wt + (size_t)(n0 + t * 16 + ln16) * Kp;
      FragBF b;
#pragma unroll
      for (int r = 0; r < 8; ++r)
        b.u[r] = *(const unsigned int*)(Wrow + k0 + kmap(r, h));
      acc[t] = __builtin_amdgcn_wmma_f32_16x16x32_bf16(
          false, a.v, false, b.v, (short)0, acc[t], false, false);
    }
  }

#pragma unroll
  for (int t = 0; t < 4; ++t) {
    int n = n0 + t * 16 + ln16;
    float bv = bias[n];
#pragma unroll
    for (int r = 0; r < 8; ++r) {
      int m = m0 + r + 8 * h;                   // C/D layout: VGPR r -> M=r (+8 for upper half)
      float v = acc[t][r] + bv;
      v = v > 0.0f ? v : 0.0f;                  // relu (every layer in the reference MLP)
      size_t o = (size_t)m * N + n;
      outBf[o] = f2bf(v);
      if (outF) outF[o] = v;
    }
  }
}

// ---------------------------------------------------------------------------
// Weight convert: f32 W[K][N] -> bf16 Wt[N][Kp] (transpose + zero pad K->Kp)
// ---------------------------------------------------------------------------
__global__ void convert_weight_kernel(const float* __restrict__ W,
                                      int K, int N, int Kp,
                                      bf16_t* __restrict__ Wt) {
  size_t tid = (size_t)blockIdx.x * blockDim.x + threadIdx.x;
  size_t total = (size_t)N * Kp;
  if (tid >= total) return;
  int k = (int)(tid % Kp);
  int n = (int)(tid / Kp);
  Wt[tid] = (k < K) ? f2bf(W[(size_t)k * N + n]) : (bf16_t)0;
}

// ---------------------------------------------------------------------------
// Farthest point sampling: one workgroup per batch.
// ---------------------------------------------------------------------------
__global__ void fps_kernel(const float* __restrict__ xyz, int stride, int N,
                           int npoint, int* __restrict__ fidx,
                           float* __restrict__ newxyz) {
  __shared__ float dist[2048];
  __shared__ float rv[256];
  __shared__ int   ri[256];
  __shared__ int   s_far;
  int b = blockIdx.x, t = threadIdx.x, nt = blockDim.x;
  const float* base = xyz + (size_t)b * N * stride;
  for (int i = t; i < N; i += nt) dist[i] = 1e10f;
  if (t == 0) s_far = 0;
  __syncthreads();
  for (int it = 0; it < npoint; ++it) {
    int far = s_far;
    if (t == 0) {
      fidx[b * npoint + it] = far;
      newxyz[((size_t)b * npoint + it) * 3 + 0] = base[(size_t)far * stride + 0];
      newxyz[((size_t)b * npoint + it) * 3 + 1] = base[(size_t)far * stride + 1];
      newxyz[((size_t)b * npoint + it) * 3 + 2] = base[(size_t)far * stride + 2];
    }
    float fx = base[(size_t)far * stride + 0];
    float fy = base[(size_t)far * stride + 1];
    float fz = base[(size_t)far * stride + 2];
    float bv = -1.0f; int bi = 0;
    for (int i = t; i < N; i += nt) {
      float dx = base[(size_t)i * stride + 0] - fx;
      float dy = base[(size_t)i * stride + 1] - fy;
      float dz = base[(size_t)i * stride + 2] - fz;
      float d = dx * dx + dy * dy + dz * dz;
      float nd = fminf(dist[i], d);
      dist[i] = nd;
      if (nd > bv) { bv = nd; bi = i; }
    }
    rv[t] = bv; ri[t] = bi;
    __syncthreads();
    for (int s = nt >> 1; s > 0; s >>= 1) {
      if (t < s) {
        if (rv[t + s] > rv[t] || (rv[t + s] == rv[t] && ri[t + s] < ri[t])) {
          rv[t] = rv[t + s]; ri[t] = ri[t + s];
        }
      }
      __syncthreads();
    }
    if (t == 0) s_far = ri[0];
    __syncthreads();
  }
}

// ---------------------------------------------------------------------------
// Ball query: first <=nsample ascending in-radius indices, pad with first.
// ---------------------------------------------------------------------------
__global__ void ballquery_kernel(const float* __restrict__ xyz, int stride, int N,
                                 const float* __restrict__ newxyz, int S,
                                 float r2, int nsample, int* __restrict__ idx,
                                 int Bn) {
  int g = blockIdx.x * blockDim.x + threadIdx.x;
  if (g >= Bn * S) return;
  int b = g / S;
  const float* base = xyz + (size_t)b * N * stride;
  float qx = newxyz[(size_t)g * 3 + 0];
  float qy = newxyz[(size_t)g * 3 + 1];
  float qz = newxyz[(size_t)g * 3 + 2];
  int* out = idx + (size_t)g * nsample;
  int cnt = 0, first = 0;
  for (int i = 0; i < N && cnt < nsample; ++i) {
    float dx = base[(size_t)i * stride + 0] - qx;
    float dy = base[(size_t)i * stride + 1] - qy;
    float dz = base[(size_t)i * stride + 2] - qz;
    if (dx * dx + dy * dy + dz * dz <= r2) {
      if (cnt == 0) first = i;
      out[cnt++] = i;
    }
  }
  for (; cnt < nsample; ++cnt) out[cnt] = first;
}

// ---------------------------------------------------------------------------
// Group + pack bf16 A matrix: rows=(b,s,j), cols=[xyz-center(3) | feat(fC) | 0pad]
// feat source is either f32 (featf) or bf16 (featb); exactly one non-null.
// ---------------------------------------------------------------------------
__global__ void group_pack_kernel(const int* __restrict__ idx,
                                  const float* __restrict__ xyz, int xstride, int Nref,
                                  const float* __restrict__ newxyz,
                                  const float* __restrict__ featf,
                                  const bf16_t* __restrict__ featb,
                                  int fstride, int fC,
                                  bf16_t* __restrict__ A,
                                  int Bn, int S, int ns, int Kp) {
  size_t tid = (size_t)blockIdx.x * blockDim.x + threadIdx.x;
  size_t total = (size_t)Bn * S * ns * Kp;
  if (tid >= total) return;
  int c = (int)(tid % Kp);
  size_t row = tid / Kp;
  size_t bs = row / ns;                 // b*S + s
  int b = (int)(bs / S);
  int p = idx[row];
  bf16_t v;
  if (c < 3) {
    float g = xyz[((size_t)b * Nref + p) * xstride + c] - newxyz[bs * 3 + c];
    v = f2bf(g);
  } else if (c < 3 + fC) {
    size_t o = ((size_t)b * Nref + p) * fstride + (c - 3);
    v = featf ? f2bf(featf[o]) : featb[o];
  } else {
    v = 0;
  }
  A[tid] = v;
}

// SA3 group_all pack: rows = b*128+j, cols = [l2x(3) | l2p(256) | pad->288]
__global__ void groupall_pack_kernel(const float* __restrict__ l2x,
                                     const bf16_t* __restrict__ l2p,
                                     bf16_t* __restrict__ A) {
  size_t tid = (size_t)blockIdx.x * blockDim.x + threadIdx.x;
  if (tid >= (size_t)512 * 288) return;
  int c = (int)(tid % 288);
  size_t row = tid / 288;
  bf16_t v;
  if (c < 3)        v = f2bf(l2x[row * 3 + c]);
  else if (c < 259) v = l2p[row * 256 + (c - 3)];
  else              v = 0;
  A[tid] = v;
}

// Max pool over nsample: in (B,S,ns,C) bf16 -> out (B,S,C) bf16
__global__ void maxpool_kernel(const bf16_t* __restrict__ in,
                               int Bn, int S, int ns, int C,
                               bf16_t* __restrict__ out) {
  size_t tid = (size_t)blockIdx.x * blockDim.x + threadIdx.x;
  size_t total = (size_t)Bn * S * C;
  if (tid >= total) return;
  int c = (int)(tid % C);
  size_t bs = tid / C;
  const bf16_t* p = in + (bs * ns) * (size_t)C + c;
  float m = -1e30f;
  for (int j = 0; j < ns; ++j) m = fmaxf(m, bf2f(p[(size_t)j * C]));
  out[tid] = f2bf(m);
}

// FP3 pack: rows = b*128+j, cols = [l2p(256) | broadcast l3p(1024)]
__global__ void fp3_pack_kernel(const bf16_t* __restrict__ l2p,
                                const bf16_t* __restrict__ l3p,
                                bf16_t* __restrict__ A) {
  size_t tid = (size_t)blockIdx.x * blockDim.x + threadIdx.x;
  if (tid >= (size_t)512 * 1280) return;
  int c = (int)(tid % 1280);
  size_t row = tid / 1280;
  int b = (int)(row >> 7);
  A[tid] = (c < 256) ? l2p[row * 256 + c] : l3p[(size_t)b * 1024 + (c - 256)];
}

// 3-NN with inverse-distance weights (stable ties -> lowest index, like top_k)
__global__ void knn3_kernel(const float* __restrict__ q, int qstride, int Nq,
                            const float* __restrict__ ref, int rstride, int Sref,
                            int Bn, int* __restrict__ oidx, float* __restrict__ ow) {
  int g = blockIdx.x * blockDim.x + threadIdx.x;
  if (g >= Bn * Nq) return;
  int b = g / Nq;
  const float* qp = q + (size_t)g * qstride;
  float qx = qp[0], qy = qp[1], qz = qp[2];
  const float* rb = ref + (size_t)b * Sref * rstride;
  float d0 = 1e30f, d1 = 1e30f, d2 = 1e30f;
  int i0 = 0, i1 = 0, i2 = 0;
  for (int i = 0; i < Sref; ++i) {
    float dx = rb[(size_t)i * rstride + 0] - qx;
    float dy = rb[(size_t)i * rstride + 1] - qy;
    float dz = rb[(size_t)i * rstride + 2] - qz;
    float d = dx * dx + dy * dy + dz * dz;
    if (d < d0)      { d2 = d1; i2 = i1; d1 = d0; i1 = i0; d0 = d; i0 = i; }
    else if (d < d1) { d2 = d1; i2 = i1; d1 = d;  i1 = i; }
    else if (d < d2) { d2 = d;  i2 = i; }
  }
  float w0 = 1.0f / (d0 + 1e-8f), w1 = 1.0f / (d1 + 1e-8f), w2 = 1.0f / (d2 + 1e-8f);
  float ws = w0 + w1 + w2;
  oidx[g * 3 + 0] = i0; oidx[g * 3 + 1] = i1; oidx[g * 3 + 2] = i2;
  ow[g * 3 + 0] = w0 / ws; ow[g * 3 + 1] = w1 / ws; ow[g * 3 + 2] = w2 / ws;
}

// FP pack: rows=(b,n) (tail rows zero-padded to Mp),
// cols = [points1(C1) | interp(C2) | 0pad -> Kp]
__global__ void fp_pack_kernel(const float* __restrict__ p1f,
                               const bf16_t* __restrict__ p1b, int p1stride, int C1,
                               const bf16_t* __restrict__ p2, int p2stride, int C2,
                               int Sref,
                               const int* __restrict__ nni,
                               const float* __restrict__ nnw,
                               bf16_t* __restrict__ A,
                               int Bn, int Nq, int Mp, int Kp) {
  size_t tid = (size_t)blockIdx.x * blockDim.x + threadIdx.x;
  size_t total = (size_t)Mp * Kp;
  if (tid >= total) return;
  int c = (int)(tid % Kp);
  size_t row = tid / Kp;
  if (row >= (size_t)Bn * Nq) { A[tid] = 0; return; }
  int b = (int)(row / Nq);
  bf16_t v;
  if (c < C1) {
    size_t o = row * p1stride + c;
    v = p1f ? f2bf(p1f[o]) : p1b[o];
  } else if (c < C1 + C2) {
    int cc = c - C1;
    const int*   ii = nni + row * 3;
    const float* ww = nnw + row * 3;
    float acc = 0.0f;
    for (int k = 0; k < 3; ++k)
      acc += ww[k] * bf2f(p2[((size_t)b * Sref + ii[k]) * p2stride + cc]);
    v = f2bf(acc);
  } else {
    v = 0;
  }
  A[tid] = v;
}

// ---------------------------------------------------------------------------
// Host orchestration
// ---------------------------------------------------------------------------
extern "C" void kernel_launch(void* const* d_in, const int* in_sizes, int n_in,
                              void* d_out, int out_size, void* d_ws, size_t ws_size,
                              hipStream_t stream) {
  (void)in_sizes; (void)n_in; (void)out_size; (void)ws_size;
  const int B = 4, N0 = 2048, C0 = 165;
  const float* x = (const float*)d_in[0];
  auto F32 = [&](int i) { return (const float*)d_in[i]; };

  char* ws = (char*)d_ws;
  size_t off = 0;
  auto alloc = [&](size_t bytes) -> char* {
    char* p = ws + off;
    off = (off + bytes + 255) & ~(size_t)255;
    return p;
  };

  bf16_t* PING = (bf16_t*)alloc((size_t)91648 * 1024 * 2);
  bf16_t* PONG = (bf16_t*)alloc((size_t)91648 * 1024 * 2);

  // Layer table: {W input index, K, N, Kp} ; bias is d_in[wi+1].
  struct LD { int wi, K, N, Kp; bf16_t* wt; };
  LD L[16] = {
    {1, 165, 1024, 192, nullptr}, {3, 1024, 512, 1024, nullptr}, {5, 512, 512, 512, nullptr},   // sa1
    {7, 515, 512, 544, nullptr},  {9, 512, 512, 512, nullptr},  {11, 512, 256, 512, nullptr},   // sa2
    {13, 259, 256, 288, nullptr}, {15, 256, 512, 256, nullptr}, {17, 512, 1024, 512, nullptr},  // sa3
    {19, 1280, 512, 1280, nullptr}, {21, 512, 512, 512, nullptr},                               // fp3
    {23, 1024, 512, 1024, nullptr}, {25, 512, 256, 512, nullptr},                               // fp2
    {27, 418, 512, 448, nullptr}, {29, 512, 256, 512, nullptr}, {31, 256, 128, 256, nullptr}    // fp1
  };
  for (int i = 0; i < 16; ++i) L[i].wt = (bf16_t*)alloc((size_t)L[i].N * L[i].Kp * 2);

  int*    FIDX1 = (int*)alloc((size_t)B * 358 * 4);
  float*  L1X   = (float*)alloc((size_t)B * 358 * 3 * 4);
  int*    IDX1  = (int*)alloc((size_t)B * 358 * 64 * 4);
  int*    FIDX2 = (int*)alloc((size_t)B * 128 * 4);
  float*  L2X   = (float*)alloc((size_t)B * 128 * 3 * 4);
  int*    IDX2  = (int*)alloc((size_t)B * 128 * 64 * 4);
  bf16_t* L1P   = (bf16_t*)alloc((size_t)B * 358 * 512 * 2);
  bf16_t* L2P   = (bf16_t*)alloc((size_t)B * 128 * 256 * 2);
  bf16_t* L3P   = (bf16_t*)alloc((size_t)B * 1024 * 2);
  bf16_t* L2P2  = (bf16_t*)alloc((size_t)512 * 512 * 2);
  bf16_t* L1P2  = (bf16_t*)alloc((size_t)1440 * 256 * 2);
  int*    NN2I  = (int*)alloc((size_t)B * 358 * 3 * 4);
  float*  NN2W  = (float*)alloc((size_t)B * 358 * 3 * 4);
  int*    NN1I  = (int*)alloc((size_t)B * 2048 * 3 * 4);
  float*  NN1W  = (float*)alloc((size_t)B * 2048 * 3 * 4);

  auto cdiv = [](long long a, long long b) { return (int)((a + b - 1) / b); };

  // Weights -> transposed bf16
  for (int i = 0; i < 16; ++i) {
    size_t tot = (size_t)L[i].N * L[i].Kp;
    convert_weight_kernel<<<cdiv((long long)tot, 256), 256, 0, stream>>>(
        F32(L[i].wi), L[i].K, L[i].N, L[i].Kp, L[i].wt);
  }

  auto gemm = [&](const bf16_t* A, int li, bf16_t* obf, float* of, int Mp) {
    long long waves = (long long)(Mp / 16) * (L[li].N / 64);
    gemm_bf16_kernel<<<cdiv(waves * 32, 256), 256, 0, stream>>>(
        A, L[li].wt, F32(L[li].wi + 1), obf, of, Mp, L[li].Kp, L[li].N);
  };

  // ---- SA1 ----
  fps_kernel<<<B, 256, 0, stream>>>(x, C0, N0, 358, FIDX1, L1X);
  ballquery_kernel<<<cdiv((long long)B * 358, 128), 128, 0, stream>>>(
      x, C0, N0, L1X, 358, 0.04f, 64, IDX1, B);
  group_pack_kernel<<<cdiv((long long)B * 358 * 64 * 192, 256), 256, 0, stream>>>(
      IDX1, x, C0, N0, L1X, x + 3, (const bf16_t*)nullptr, C0, 162,
      PING, B, 358, 64, 192);
  gemm(PING, 0, PONG, nullptr, 91648);
  gemm(PONG, 1, PING, nullptr, 91648);
  gemm(PING, 2, PONG, nullptr, 91648);
  maxpool_kernel<<<cdiv((long long)B * 358 * 512, 256), 256, 0, stream>>>(
      PONG, B, 358, 64, 512, L1P);

  // ---- SA2 ----
  fps_kernel<<<B, 256, 0, stream>>>(L1X, 3, 358, 128, FIDX2, L2X);
  ballquery_kernel<<<cdiv((long long)B * 128, 128), 128, 0, stream>>>(
      L1X, 3, 358, L2X, 128, 0.16f, 64, IDX2, B);
  group_pack_kernel<<<cdiv((long long)B * 128 * 64 * 544, 256), 256, 0, stream>>>(
      IDX2, L1X, 3, 358, L2X, (const float*)nullptr, L1P, 512, 512,
      PING, B, 128, 64, 544);
  gemm(PING, 3, PONG, nullptr, 32768);
  gemm(PONG, 4, PING, nullptr, 32768);
  gemm(PING, 5, PONG, nullptr, 32768);
  maxpool_kernel<<<cdiv((long long)B * 128 * 256, 256), 256, 0, stream>>>(
      PONG, B, 128, 64, 256, L2P);

  // ---- SA3 (group_all) ----
  groupall_pack_kernel<<<cdiv((long long)512 * 288, 256), 256, 0, stream>>>(L2X, L2P, PING);
  gemm(PING, 6, PONG, nullptr, 512);
  gemm(PONG, 7, PING, nullptr, 512);
  gemm(PING, 8, PONG, nullptr, 512);
  maxpool_kernel<<<cdiv((long long)B * 1024, 256), 256, 0, stream>>>(
      PONG, B, 1, 128, 1024, L3P);

  // ---- FP3 ----
  fp3_pack_kernel<<<cdiv((long long)512 * 1280, 256), 256, 0, stream>>>(L2P, L3P, PING);
  gemm(PING, 9, PONG, nullptr, 512);
  gemm(PONG, 10, L2P2, nullptr, 512);

  // ---- FP2 ----
  knn3_kernel<<<cdiv((long long)B * 358, 128), 128, 0, stream>>>(
      L1X, 3, 358, L2X, 3, 128, B, NN2I, NN2W);
  fp_pack_kernel<<<cdiv((long long)1440 * 1024, 256), 256, 0, stream>>>(
      (const float*)nullptr, L1P, 512, 512, L2P2, 512, 512, 128,
      NN2I, NN2W, PING, B, 358, 1440, 1024);
  gemm(PING, 11, PONG, nullptr, 1440);
  gemm(PONG, 12, L1P2, nullptr, 1440);

  // ---- FP1 ----
  knn3_kernel<<<cdiv((long long)B * 2048, 128), 128, 0, stream>>>(
      x, C0, 2048, L1X, 3, 358, B, NN1I, NN1W);
  fp_pack_kernel<<<cdiv((long long)8192 * 448, 256), 256, 0, stream>>>(
      x + 3, (const bf16_t*)nullptr, C0, 162, L1P2, 256, 256, 358,
      NN1I, NN1W, PING, B, 2048, 8192, 448);
  gemm(PING, 13, PONG, nullptr, 8192);
  gemm(PONG, 14, PING, nullptr, 8192);
  gemm(PING, 15, PONG, (float*)d_out, 8192);   // final f32 output (4,2048,128)
}
